// Decoder_43860206026877
// MI455X (gfx1250) — matrix operand order repaired and checked by reading, not compiled, not run on hardware
//
#include <hip/hip_runtime.h>
#include <stdint.h>
#include <stddef.h>

// ---------------- problem constants ----------------
#define BT   2
#define SEQ  2048
#define DM   512
#define HD   8
#define RD   64
#define FF   2048
#define NT   (BT*SEQ)          // 4096 tokens

typedef __attribute__((ext_vector_type(16))) __bf16 v16bf;
typedef __attribute__((ext_vector_type(8)))  float  v8f;
typedef __attribute__((ext_vector_type(4)))  uint32_t v4u;
typedef __attribute__((ext_vector_type(8)))  uint32_t v8u;

struct Frag { union { uint32_t u[8]; v16bf v; }; };

__device__ __forceinline__ uint16_t f2bf(float f) {
    uint32_t u = __float_as_uint(f);
    u += 0x7FFFu + ((u >> 16) & 1u);      // round-to-nearest-even
    return (uint16_t)(u >> 16);
}

// A-fragment K offset for 16-bit 16x32 A (contiguous order, matches B):
// lanes 0-15: V0..7 -> K=0..15 ; lanes 16-31: K=16..31
__device__ __forceinline__ int koffA(int j, int half) {
    return 2*j + 16*half;
}

__device__ __forceinline__ v8f wmma_bf(const Frag& a, const Frag& b, v8f c) {
    return __builtin_amdgcn_wmma_f32_16x16x32_bf16(
        false, a.v, false, b.v, (short)0, c, false, false);
}

// ---- Tensor Data Mover: DMA a [rows=16] x [64 dwords] tile into LDS ----
// D# built per CDNA5 ISA (2D tensor, data_size=4B,
// groups 2/3 disabled).  Tracked by TENSORcnt.
__device__ __forceinline__ void tdm_load_tile16x64(const uint32_t* gsrc,
                                                   uint32_t lds_off,
                                                   uint32_t N_dw) {
    uint64_t ga = (uint64_t)(uintptr_t)gsrc;
    v4u g0;
    g0[0] = 1u;                                   // count=1, user mode
    g0[1] = lds_off;                              // LDS byte address
    g0[2] = (uint32_t)ga;                         // global_addr[31:0]
    g0[3] = ((uint32_t)(ga >> 32) & 0x01FFFFFFu)  // global_addr[56:32]
            | (2u << 30);                         // type=2 ("image")
    v8u g1;
    g1[0] = (2u << 16);                           // wg_mask=0, data_size=4B
    g1[1] = (N_dw & 0xFFFFu) << 16;               // tensor_dim0[15:0]
    g1[2] = (N_dw >> 16) | (16u << 16);           // tensor_dim0[31:16], tensor_dim1=16
    g1[3] = (64u << 16);                          // tensor_dim1 hi=0, tile_dim0=64
    g1[4] = 16u;                                  // tile_dim1=16, tile_dim2=0
    g1[5] = N_dw;                                 // tensor_dim0_stride[31:0]
    g1[6] = 0u;                                   // stride0 hi, stride1 lo
    g1[7] = 0u;
    asm volatile("tensor_load_to_lds %0, %1" :: "s"(g0), "s"(g1) : "memory");
}

// ---------------- conversion / packing ----------------
__global__ void cvt_bf16_kernel(const float* __restrict__ src,
                                uint16_t* __restrict__ dst, int n) {
    for (int i = blockIdx.x*blockDim.x + threadIdx.x; i < n;
         i += gridDim.x*blockDim.x)
        dst[i] = f2bf(src[i]);
}

// W[K,N] fp32 -> packed u32[(K/2)*N] = {bf16(W[2k,n]), bf16(W[2k+1,n])}
__global__ void pack_kn_kernel(const float* __restrict__ W,
                               uint32_t* __restrict__ dst, int K, int N) {
    int n = (K >> 1) * N;
    for (int i = blockIdx.x*blockDim.x + threadIdx.x; i < n;
         i += gridDim.x*blockDim.x) {
        int kp = i / N, c = i % N;
        uint32_t lo = f2bf(W[(size_t)(2*kp)*N + c]);
        uint32_t hi = f2bf(W[(size_t)(2*kp+1)*N + c]);
        dst[i] = lo | (hi << 16);
    }
}

// W[H,D,R] fp32 -> packed u32[(D/2)*(H*R)] with flat col = h*R + r
__global__ void pack_hdr_kernel(const float* __restrict__ W,
                                uint32_t* __restrict__ dst) {
    int n = (DM >> 1) * DM;   // (D/2) * (H*R) ; H*R == DM
    for (int i = blockIdx.x*blockDim.x + threadIdx.x; i < n;
         i += gridDim.x*blockDim.x) {
        int dp = i / DM, c = i % DM;
        int h = c >> 6, r = c & 63;
        size_t base = (size_t)h * DM * RD;
        uint32_t lo = f2bf(W[base + (size_t)(2*dp)  *RD + r]);
        uint32_t hi = f2bf(W[base + (size_t)(2*dp+1)*RD + r]);
        dst[i] = lo | (hi << 16);
    }
}

// bf16[T,DM] -> row-pair packed u32[(T/2)*DM]  (for V as B-fragment)
__global__ void repack_pairs_kernel(const uint16_t* __restrict__ src,
                                    uint32_t* __restrict__ dst, int Tn) {
    int n = (Tn >> 1) * DM;
    for (int i = blockIdx.x*blockDim.x + threadIdx.x; i < n;
         i += gridDim.x*blockDim.x) {
        int tp = i / DM, c = i % DM;
        uint32_t lo = src[(size_t)(2*tp)*DM + c];
        uint32_t hi = src[(size_t)(2*tp+1)*DM + c];
        dst[i] = lo | (hi << 16);
    }
}

// ---------------- bf16 WMMA GEMM with TDM-staged weights ----------------
// Block = 4 waves = 64x64 output tile; wave w owns rows [m0+16w, m0+16w+16).
// Weight K-slice (16 packed rows x 64 dwords) DMA'd to LDS, double-buffered.
template<bool RELU, bool RESID, bool OUTBF, bool OUTF>
__global__ __launch_bounds__(128)
void gemm_bf16_kernel(const uint16_t* __restrict__ A,
                      const uint32_t* __restrict__ Wp,
                      const float*    __restrict__ bias,
                      const float*    __restrict__ resid,
                      uint16_t*       __restrict__ outbf,
                      float*          __restrict__ outf,
                      int M, int N, int K, float scale) {
    __shared__ uint32_t wtile[2][16*64];
    int wave = threadIdx.x >> 5;
    int lane = threadIdx.x & 31;
    int ntn  = N >> 6;
    int m0   = (blockIdx.x / ntn) * 64 + wave * 16;
    int n0   = (blockIdx.x % ntn) * 64;
    int m = lane & 15, half = lane >> 4;
    int ksteps = K >> 5;

    v8f acc[4];
    #pragma unroll
    for (int t = 0; t < 4; ++t) acc[t] = {};

    // prologue: DMA first weight slice
    if (wave == 0)
        tdm_load_tile16x64(Wp + (size_t)0*N + n0,
                           (uint32_t)(uintptr_t)&wtile[0][0], (uint32_t)N);

    const uint16_t* arow = A + (size_t)(m0 + m) * K;
    for (int i = 0; i < ksteps; ++i) {
        int k   = i << 5;
        int cur = i & 1;
        if (wave == 0) {
            if (i + 1 < ksteps) {
                tdm_load_tile16x64(Wp + (size_t)((k >> 1) + 16)*N + n0,
                                   (uint32_t)(uintptr_t)&wtile[cur ^ 1][0],
                                   (uint32_t)N);
                // slice i complete (in-order TENSORcnt; next DMA outstanding)
                __builtin_amdgcn_s_wait_tensorcnt((short)1);
            } else {
                __builtin_amdgcn_s_wait_tensorcnt((short)0);
            }
        }
        __syncthreads();

        Frag a;
        #pragma unroll
        for (int j = 0; j < 8; ++j)
            a.u[j] = *(const uint32_t*)(arow + k + koffA(j, half));
        #pragma unroll
        for (int nt = 0; nt < 4; ++nt) {
            Frag b;
            #pragma unroll
            for (int j = 0; j < 8; ++j)
                b.u[j] = wtile[cur][(j + 8*half)*64 + nt*16 + m];
            acc[nt] = wmma_bf(a, b, acc[nt]);
        }
        __syncthreads();   // protect buffer reuse before next DMA overwrite
    }

    #pragma unroll
    for (int nt = 0; nt < 4; ++nt) {
        int col = n0 + nt*16 + m;
        float bv = bias[col];
        #pragma unroll
        for (int j = 0; j < 8; ++j) {
            int row = m0 + j + 8*half;
            float v = (acc[nt][j] + bv) * scale;
            if (RELU)  v = fmaxf(v, 0.f);
            if (RESID) v += resid[(size_t)row*N + col];
            if (OUTF)  outf [(size_t)row*N + col] = v;
            if (OUTBF) outbf[(size_t)row*N + col] = f2bf(v);
        }
    }
}

// ---------------- flash attention (no mask) ----------------
// one wave handles one (b,h, 16-query block); 32 keys per iteration
__global__ __launch_bounds__(128)
void attn_kernel(const uint16_t* __restrict__ Q,
                 const uint16_t* __restrict__ Kt,
                 const uint32_t* __restrict__ Vp,
                 uint16_t*       __restrict__ O) {
    __shared__ uint16_t lds[4][16*32];
    int wave = threadIdx.x >> 5;
    int lane = threadIdx.x & 31;
    int wid  = blockIdx.x*4 + wave;
    const int QB = SEQ/16;
    int qb = wid % QB;
    int t  = wid / QB;
    int h  = t % HD;
    int b  = t / HD;
    int m = lane & 15, half = lane >> 4;
    int cbase = h * RD;

    // Q as two A-fragments (r=0..31, r=32..63)
    size_t rowQ = (size_t)(b*SEQ + qb*16 + m) * DM + cbase;
    Frag qa0, qa1;
    #pragma unroll
    for (int j = 0; j < 8; ++j) {
        qa0.u[j] = *(const uint32_t*)(Q + rowQ + koffA(j, half));
        qa1.u[j] = *(const uint32_t*)(Q + rowQ + 32 + koffA(j, half));
    }

    v8f o0 = {}, o1 = {}, o2 = {}, o3 = {};
    float rmax[8], rsum[8];
    #pragma unroll
    for (int j = 0; j < 8; ++j) { rmax[j] = -1e30f; rsum[j] = 0.f; }

    for (int kb = 0; kb < SEQ/32; ++kb) {
        int t0 = kb * 32;
        // scores: S[16 queries][32 keys] as two f32 fragments
        v8f s0 = {}, s1 = {};
        {
            Frag kf;
            size_t rk0 = (size_t)(b*SEQ + t0 + m) * DM + cbase + 16*half;
            #pragma unroll
            for (int j = 0; j < 8; ++j)
                kf.u[j] = *(const uint32_t*)(Kt + rk0 + 2*j);
            s0 = wmma_bf(qa0, kf, s0);
            #pragma unroll
            for (int j = 0; j < 8; ++j)
                kf.u[j] = *(const uint32_t*)(Kt + rk0 + 32 + 2*j);
            s0 = wmma_bf(qa1, kf, s0);
            size_t rk1 = (size_t)(b*SEQ + t0 + 16 + m) * DM + cbase + 16*half;
            #pragma unroll
            for (int j = 0; j < 8; ++j)
                kf.u[j] = *(const uint32_t*)(Kt + rk1 + 2*j);
            s1 = wmma_bf(qa0, kf, s1);
            #pragma unroll
            for (int j = 0; j < 8; ++j)
                kf.u[j] = *(const uint32_t*)(Kt + rk1 + 32 + 2*j);
            s1 = wmma_bf(qa1, kf, s1);
        }
        // online softmax over the 32 new columns (row = j + 8*half)
        #pragma unroll
        for (int j = 0; j < 8; ++j) {
            float mx = fmaxf(s0[j], s1[j]);
            #pragma unroll
            for (int w = 1; w < 16; w <<= 1) mx = fmaxf(mx, __shfl_xor(mx, w, 32));
            float nm  = fmaxf(rmax[j], mx);
            float fct = __expf(rmax[j] - nm);
            rmax[j] = nm;
            float p0 = __expf(s0[j] - nm);
            float p1 = __expf(s1[j] - nm);
            float rs = p0 + p1;
            #pragma unroll
            for (int w = 1; w < 16; w <<= 1) rs += __shfl_xor(rs, w, 32);
            rsum[j] = rsum[j]*fct + rs;
            o0[j] *= fct; o1[j] *= fct; o2[j] *= fct; o3[j] *= fct;
            lds[wave][(j + 8*half)*32 + m]      = f2bf(p0);
            lds[wave][(j + 8*half)*32 + 16 + m] = f2bf(p1);
        }
        __syncthreads();
        // re-load P as A-fragment (16x32)
        Frag pa;
        #pragma unroll
        for (int j = 0; j < 8; ++j)
            pa.u[j] = *(const uint32_t*)(&lds[wave][m*32 + koffA(j, half)]);
        // V as B-fragments from pair-packed layout
        const uint32_t* vp = Vp + (size_t)(((b*SEQ + t0) >> 1) + 8*half) * DM
                                + cbase + m;
        Frag vf;
        #pragma unroll
        for (int j = 0; j < 8; ++j) vf.u[j] = vp[(size_t)j*DM + 0*16];
        o0 = wmma_bf(pa, vf, o0);
        #pragma unroll
        for (int j = 0; j < 8; ++j) vf.u[j] = vp[(size_t)j*DM + 1*16];
        o1 = wmma_bf(pa, vf, o1);
        #pragma unroll
        for (int j = 0; j < 8; ++j) vf.u[j] = vp[(size_t)j*DM + 2*16];
        o2 = wmma_bf(pa, vf, o2);
        #pragma unroll
        for (int j = 0; j < 8; ++j) vf.u[j] = vp[(size_t)j*DM + 3*16];
        o3 = wmma_bf(pa, vf, o3);
        __syncthreads();
    }
    // normalize and store bf16
    #pragma unroll
    for (int j = 0; j < 8; ++j) {
        float inv = 1.f / rsum[j];
        size_t ro = (size_t)(b*SEQ + qb*16 + j + 8*half) * DM + cbase + m;
        O[ro + 0]  = f2bf(o0[j] * inv);
        O[ro + 16] = f2bf(o1[j] * inv);
        O[ro + 32] = f2bf(o2[j] * inv);
        O[ro + 48] = f2bf(o3[j] * inv);
    }
}

// ---------------- LayerNorm (row = DM, eps = 1e-3) ----------------
__global__ __launch_bounds__(128)
void ln_kernel(const float* __restrict__ in,
               const float* __restrict__ g, const float* __restrict__ bt,
               float* __restrict__ outf, uint16_t* __restrict__ outbf) {
    __shared__ float s1[4], s2[4];
    int row = blockIdx.x;
    const float* x = in + (size_t)row * DM;
    float v[4], sum = 0.f, sq = 0.f;
    #pragma unroll
    for (int i = 0; i < 4; ++i) {
        v[i] = x[threadIdx.x + i*128];
        sum += v[i]; sq += v[i]*v[i];
    }
    #pragma unroll
    for (int w = 1; w < 32; w <<= 1) {
        sum += __shfl_xor(sum, w, 32);
        sq  += __shfl_xor(sq,  w, 32);
    }
    int wave = threadIdx.x >> 5;
    if ((threadIdx.x & 31) == 0) { s1[wave] = sum; s2[wave] = sq; }
    __syncthreads();
    sum = s1[0] + s1[1] + s1[2] + s1[3];
    sq  = s2[0] + s2[1] + s2[2] + s2[3];
    float mean = sum * (1.f/DM);
    float var  = sq * (1.f/DM) - mean*mean;
    float rstd = rsqrtf(var + 1e-3f);
    #pragma unroll
    for (int i = 0; i < 4; ++i) {
        int c = threadIdx.x + i*128;
        float y = (v[i] - mean) * rstd * g[c] + bt[c];
        if (outf)  outf [(size_t)row*DM + c] = y;
        if (outbf) outbf[(size_t)row*DM + c] = f2bf(y);
    }
}

// ---------------- host orchestration ----------------
extern "C" void kernel_launch(void* const* d_in, const int* in_sizes, int n_in,
                              void* d_out, int out_size, void* d_ws, size_t ws_size,
                              hipStream_t stream) {
    (void)in_sizes; (void)n_in; (void)out_size; (void)ws_size;
    const float* x   = (const float*)d_in[0];
    const float* ctx = (const float*)d_in[1];
    const float* Wq1 = (const float*)d_in[2];  const float* bq1 = (const float*)d_in[3];
    const float* Wk1 = (const float*)d_in[4];  const float* bk1 = (const float*)d_in[5];
    const float* Wv1 = (const float*)d_in[6];  const float* bv1 = (const float*)d_in[7];
    const float* Wo1 = (const float*)d_in[8];  const float* bo1 = (const float*)d_in[9];
    const float* Wq2 = (const float*)d_in[10]; const float* bq2 = (const float*)d_in[11];
    const float* Wk2 = (const float*)d_in[12]; const float* bk2 = (const float*)d_in[13];
    const float* Wv2 = (const float*)d_in[14]; const float* bv2 = (const float*)d_in[15];
    const float* Wo2 = (const float*)d_in[16]; const float* bo2 = (const float*)d_in[17];
    const float* lng = (const float*)d_in[18]; const float* lnb = (const float*)d_in[19];
    const float* W1  = (const float*)d_in[20]; const float* b1  = (const float*)d_in[21];
    const float* W2  = (const float*)d_in[22]; const float* b2  = (const float*)d_in[23];
    float* out = (float*)d_out;

    // workspace carve-up
    char* base = (char*)d_ws;
    size_t off = 0;
    auto alloc = [&](size_t bytes) -> char* {
        char* p = base + off;
        off += (bytes + 255) & ~(size_t)255;
        return p;
    };
    uint16_t* xbf   = (uint16_t*)alloc((size_t)NT*DM*2);
    uint16_t* cbf   = (uint16_t*)alloc((size_t)NT*DM*2);
    uint32_t* wq1p  = (uint32_t*)alloc((size_t)(DM/2)*DM*4);
    uint32_t* wk1p  = (uint32_t*)alloc((size_t)(DM/2)*DM*4);
    uint32_t* wv1p  = (uint32_t*)alloc((size_t)(DM/2)*DM*4);
    uint32_t* wo1p  = (uint32_t*)alloc((size_t)(DM/2)*DM*4);
    uint32_t* wq2p  = (uint32_t*)alloc((size_t)(DM/2)*DM*4);
    uint32_t* wk2p  = (uint32_t*)alloc((size_t)(DM/2)*DM*4);
    uint32_t* wv2p  = (uint32_t*)alloc((size_t)(DM/2)*DM*4);
    uint32_t* wo2p  = (uint32_t*)alloc((size_t)(DM/2)*DM*4);
    uint32_t* w1p   = (uint32_t*)alloc((size_t)(DM/2)*FF*4);
    uint32_t* w2p   = (uint32_t*)alloc((size_t)(FF/2)*DM*4);
    uint16_t* qbf   = (uint16_t*)alloc((size_t)NT*DM*2);
    uint16_t* kbf   = (uint16_t*)alloc((size_t)NT*DM*2);
    uint16_t* vbf   = (uint16_t*)alloc((size_t)NT*DM*2);
    uint32_t* vpack = (uint32_t*)alloc((size_t)(NT/2)*DM*4);
    uint16_t* obf   = (uint16_t*)alloc((size_t)NT*DM*2);
    float*    sum   = (float*)   alloc((size_t)NT*DM*4);
    float*    xcur  = (float*)   alloc((size_t)NT*DM*4);
    uint16_t* xcurb = (uint16_t*)alloc((size_t)NT*DM*2);
    uint16_t* hbf   = (uint16_t*)alloc((size_t)NT*FF*2);

    // GEMM launchers (block = 64x64 tile, 128 threads)
    auto gemm_bf = [&](const uint16_t* A, const uint32_t* Wp, const float* bias,
                       uint16_t* ob, int M, int N, int K, float scale) {
        gemm_bf16_kernel<false,false,true,false>
            <<<(M/64)*(N/64), 128, 0, stream>>>(A, Wp, bias, nullptr, ob,
                                                nullptr, M, N, K, scale);
    };
    auto gemm_relu = [&](const uint16_t* A, const uint32_t* Wp, const float* bias,
                         uint16_t* ob, int M, int N, int K) {
        gemm_bf16_kernel<true,false,true,false>
            <<<(M/64)*(N/64), 128, 0, stream>>>(A, Wp, bias, nullptr, ob,
                                                nullptr, M, N, K, 1.f);
    };
    auto gemm_res = [&](const uint16_t* A, const uint32_t* Wp, const float* bias,
                        const float* resid, float* of, int M, int N, int K) {
        gemm_bf16_kernel<false,true,false,true>
            <<<(M/64)*(N/64), 128, 0, stream>>>(A, Wp, bias, resid, nullptr,
                                                of, M, N, K, 1.f);
    };

    // ---- convert + pack ----
    cvt_bf16_kernel<<<1024, 256, 0, stream>>>(x,   xbf, NT*DM);
    cvt_bf16_kernel<<<1024, 256, 0, stream>>>(ctx, cbf, NT*DM);
    pack_hdr_kernel<<<512, 256, 0, stream>>>(Wq1, wq1p);
    pack_hdr_kernel<<<512, 256, 0, stream>>>(Wk1, wk1p);
    pack_hdr_kernel<<<512, 256, 0, stream>>>(Wv1, wv1p);
    pack_hdr_kernel<<<512, 256, 0, stream>>>(Wq2, wq2p);
    pack_hdr_kernel<<<512, 256, 0, stream>>>(Wk2, wk2p);
    pack_hdr_kernel<<<512, 256, 0, stream>>>(Wv2, wv2p);
    pack_kn_kernel<<<512,  256, 0, stream>>>(Wo1, wo1p, DM, DM);
    pack_kn_kernel<<<512,  256, 0, stream>>>(Wo2, wo2p, DM, DM);
    pack_kn_kernel<<<2048, 256, 0, stream>>>(W1,  w1p,  DM, FF);
    pack_kn_kernel<<<2048, 256, 0, stream>>>(W2,  w2p,  FF, DM);

    const float qscale = 0.125f;   // 1/sqrt(R=64)

    // ---- self-attention block ----
    gemm_bf(xbf, wq1p, bq1, qbf, NT, DM, DM, qscale);
    gemm_bf(xbf, wk1p, bk1, kbf, NT, DM, DM, 1.f);
    gemm_bf(xbf, wv1p, bv1, vbf, NT, DM, DM, 1.f);
    repack_pairs_kernel<<<1024, 256, 0, stream>>>(vbf, vpack, NT);
    attn_kernel<<<(BT*HD*(SEQ/16))/4, 128, 0, stream>>>(qbf, kbf, vpack, obf);
    gemm_res(obf, wo1p, bo1, x, sum, NT, DM, DM);
    ln_kernel<<<NT, 128, 0, stream>>>(sum, lng, lnb, xcur, xcurb);

    // ---- cross-attention block ----
    gemm_bf(xcurb, wq2p, bq2, qbf, NT, DM, DM, qscale);
    gemm_bf(cbf,   wk2p, bk2, kbf, NT, DM, DM, 1.f);
    gemm_bf(cbf,   wv2p, bv2, vbf, NT, DM, DM, 1.f);
    repack_pairs_kernel<<<1024, 256, 0, stream>>>(vbf, vpack, NT);
    attn_kernel<<<(BT*HD*(SEQ/16))/4, 128, 0, stream>>>(qbf, kbf, vpack, obf);
    gemm_res(obf, wo2p, bo2, xcur, sum, NT, DM, DM);
    ln_kernel<<<NT, 128, 0, stream>>>(sum, lng, lnb, xcur, xcurb);

    // ---- FFN ----
    gemm_relu(xcurb, w1p, b1, hbf, NT, FF, DM);
    gemm_res(hbf, w2p, b2, xcur, sum, NT, DM, FF);
    ln_kernel<<<NT, 128, 0, stream>>>(sum, lng, lnb, out, nullptr);
}
